// BiLSTM_CRF_10110353015061
// MI455X (gfx1250) — matrix operand order repaired and checked by reading, not compiled, not run on hardware
//
#include <hip/hip_runtime.h>
#include <hip/hip_bf16.h>
#include <stdint.h>

typedef __attribute__((ext_vector_type(16))) _Float16 v16h;
typedef __attribute__((ext_vector_type(8)))  float    v8f;

#define TAGS      32
#define START_TAG 30
#define STOP_TAG  31
#define NEGV      (-10000.0f)
#define LOG2E     1.4426950408889634f
#define LN2       0.6931471805599453f
#define NC        2            // independent 16-row recurrence chains per wave

// One wave owns NC*16 batch rows for the entire scan, base-2 log domain:
//   p = exp2(alpha2 - rowmax2); s = p * exp2(trans2)^T (WMMA); alpha2' = log2(s)
// (rowmax2 accumulates in Macc2; emissions added lazily next iteration).
// Working register layout == WMMA 16-bit A layout:
//   lanes 0..15  own row=lane,    tags {0..7, 16..23}
//   lanes 16..31 own row=lane-16, tags {8..15, 24..31}
__global__ __launch_bounds__(32)
void crf_nll_kernel(const float* __restrict__ feats,
                    const float* __restrict__ trans,
                    const int*   __restrict__ tags,
                    float* __restrict__ out,
                    int B, int L)
{
    __shared__ __align__(16) float ldsAlpha[NC * 16 * TAGS];  // per-chain [row][tag] log2(s)

    const int  lane    = threadIdx.x;
    const int  row     = lane & 15;
    const bool hi      = lane >= 16;
    const int  tbase   = hi ? 8 : 0;
    const int  rowBase = blockIdx.x * (NC * 16);

    // ---- B matrices: exp2(trans * log2e), WMMA 16-bit B layout (loop-constant) ----
    v16h Bm0, Bm1;
    {
        const int n0 = lane & 15;
        const int n1 = 16 + (lane & 15);
        const int kb = hi ? 8 : 0;
        #pragma unroll
        for (int e = 0; e < 16; ++e) {
            const int k = kb + (e < 8 ? e : e + 8);
            Bm0[e] = (_Float16)__builtin_amdgcn_exp2f(trans[n0 * TAGS + k] * LOG2E);
            Bm1[e] = (_Float16)__builtin_amdgcn_exp2f(trans[n1 * TAGS + k] * LOG2E);
        }
    }

    // ---- init alpha2: NEG*log2e everywhere, 0 at START ----
    for (int idx = lane; idx < NC * 16 * TAGS; idx += 32)
        ldsAlpha[idx] = ((idx & 31) == START_TAG) ? 0.0f : (NEGV * LOG2E);
    __syncthreads();

    float featbuf[NC][16];                                // raw emissions of step l-1
    #pragma unroll
    for (int c = 0; c < NC; ++c)
        #pragma unroll
        for (int i = 0; i < 16; ++i) featbuf[c][i] = 0.0f;

    float Macc2[NC], gold[NC];
    int   prevTag[NC], tgCur[NC];
    const float* featW[NC];
    const float* featG[NC];
    const int*   tagR[NC];

    const size_t rowStride = (size_t)L * TAGS;
    #pragma unroll
    for (int c = 0; c < NC; ++c) {
        Macc2[c] = 0.0f; gold[c] = 0.0f; prevTag[c] = START_TAG;
        featW[c] = feats + (size_t)(rowBase + 16*c + row) * rowStride + tbase;
        featG[c] = feats + (size_t)(rowBase + 16*c + row) * rowStride;
        tagR[c]  = tags  + (size_t)(rowBase + 16*c + row) * L;
        tgCur[c] = hi ? START_TAG : tagR[c][0];           // tags[l=0], preloaded
    }

    for (int l = 0; l < L; ++l) {
        // ---- gold: issue loads early (tag for l+1; gathers for step l use tgCur) ----
        int   tgNext[NC];
        float gfeat[NC], gtrans[NC];
        if (!hi) {
            const int ln = (l + 1 < L) ? (l + 1) : l;
            #pragma unroll
            for (int c = 0; c < NC; ++c) {
                tgNext[c] = tagR[c][ln];
                gfeat[c]  = featG[c][(size_t)l * TAGS + tgCur[c]];
                gtrans[c] = trans[tgCur[c] * TAGS + prevTag[c]];
            }
        }

        // ---- working phase + emission double-buffer + exp/pack + WMMA, per chain ----
        v8f cc0[NC], cc1[NC];
        #pragma unroll
        for (int c = 0; c < NC; ++c) {
            float v[16];
            #pragma unroll
            for (int blk = 0; blk < 2; ++blk) {           // tag blocks +0 / +16
                const int o = blk * 8;
                #pragma unroll
                for (int k4 = 0; k4 < 2; ++k4) {
                    const float4 q = *(const float4*)
                        &ldsAlpha[c * 16 * TAGS + row * TAGS + tbase + 16*blk + 4*k4];
                    v[o+4*k4+0] = __builtin_fmaf(featbuf[c][o+4*k4+0], LOG2E, q.x);
                    v[o+4*k4+1] = __builtin_fmaf(featbuf[c][o+4*k4+1], LOG2E, q.y);
                    v[o+4*k4+2] = __builtin_fmaf(featbuf[c][o+4*k4+2], LOG2E, q.z);
                    v[o+4*k4+3] = __builtin_fmaf(featbuf[c][o+4*k4+3], LOG2E, q.w);
                }
            }

            // fetch raw emissions of step l (consumed next iteration)
            {
                const float* p = featW[c] + (size_t)l * TAGS;
                #pragma unroll
                for (int blk = 0; blk < 2; ++blk) {
                    const int o = blk * 8;
                    #pragma unroll
                    for (int k4 = 0; k4 < 2; ++k4) {
                        const float4 q = *(const float4*)(p + 16*blk + 4*k4);
                        featbuf[c][o+4*k4+0] = q.x; featbuf[c][o+4*k4+1] = q.y;
                        featbuf[c][o+4*k4+2] = q.z; featbuf[c][o+4*k4+3] = q.w;
                    }
                }
                const int lp = (l + 2 < L) ? (l + 2) : (L - 1);
                __builtin_prefetch(featW[c] + (size_t)lp * TAGS, 0, 0);
            }

            // row max over all 32 tags
            float m = v[0];
            #pragma unroll
            for (int i = 1; i < 16; ++i) m = fmaxf(m, v[i]);
            m = fmaxf(m, __shfl_xor(m, 16));
            Macc2[c] += m;

            // A = exp2(v - m) packed to f16 (working layout == A layout)
            union { v16h vh; uint32_t u[8]; } A;
            #pragma unroll
            for (int j = 0; j < 8; ++j) {
                const float p0 = __builtin_amdgcn_exp2f(v[2*j+0] - m);
                const float p1 = __builtin_amdgcn_exp2f(v[2*j+1] - m);
                A.u[j] = __builtin_bit_cast(uint32_t, __builtin_amdgcn_cvt_pkrtz(p0, p1));
            }

            const v8f zero = {0.f,0.f,0.f,0.f,0.f,0.f,0.f,0.f};
            cc0[c] = __builtin_amdgcn_wmma_f32_16x16x32_f16(false, A.vh, false, Bm0,
                                                            (short)0, zero, false, false);
            cc1[c] = __builtin_amdgcn_wmma_f32_16x16x32_f16(false, A.vh, false, Bm1,
                                                            (short)0, zero, false, false);
        }

        // ---- gold: accumulate (loads issued at top of iteration, latency covered) ----
        if (!hi) {
            #pragma unroll
            for (int c = 0; c < NC; ++c) {
                gold[c]   += gfeat[c] + gtrans[c];
                prevTag[c] = tgCur[c];
                tgCur[c]   = tgNext[c];
            }
        }

        __syncthreads();   // 1-wave WG: pure dscnt wait; orders read-before-write

        // ---- scatter log2(s) from D layout to canonical LDS ----
        const int off = hi ? 8 : 0;
        const int col = lane & 15;
        #pragma unroll
        for (int c = 0; c < NC; ++c)
            #pragma unroll
            for (int r = 0; r < 8; ++r) {
                ldsAlpha[c * 16 * TAGS + (r + off) * TAGS + col     ] =
                    __builtin_amdgcn_logf(cc0[c][r]);
                ldsAlpha[c * 16 * TAGS + (r + off) * TAGS + col + 16] =
                    __builtin_amdgcn_logf(cc1[c][r]);
            }
        __syncthreads();
    }

    // ---- final: alpha2 = log2S + (feat_{L-1} + trans[STOP]) * log2e + Macc2 ----
    #pragma unroll
    for (int c = 0; c < NC; ++c) {
        float vf[16];
        #pragma unroll
        for (int blk = 0; blk < 2; ++blk) {
            const int o = blk * 8;
            #pragma unroll
            for (int k4 = 0; k4 < 2; ++k4) {
                const float4 q = *(const float4*)
                    &ldsAlpha[c * 16 * TAGS + row * TAGS + tbase + 16*blk + 4*k4];
                #pragma unroll
                for (int e = 0; e < 4; ++e) {
                    const int   i  = o + 4*k4 + e;
                    const int   tg = tbase + 16*blk + 4*k4 + e;
                    const float qv = (e==0) ? q.x : (e==1) ? q.y : (e==2) ? q.z : q.w;
                    vf[i] = __builtin_fmaf(featbuf[c][i] + trans[STOP_TAG * TAGS + tg],
                                           LOG2E, qv);
                }
            }
        }
        float m = vf[0];
        #pragma unroll
        for (int i = 1; i < 16; ++i) m = fmaxf(m, vf[i]);
        const float M2 = fmaxf(m, __shfl_xor(m, 16));
        float s = 0.0f;
        #pragma unroll
        for (int i = 0; i < 16; ++i) s += __builtin_amdgcn_exp2f(vf[i] - M2);
        s += __shfl_xor(s, 16);

        if (!hi) {
            const float g = gold[c] + trans[STOP_TAG * TAGS + prevTag[c]];
            out[rowBase + 16*c + row] =
                (M2 + __builtin_amdgcn_logf(s) + Macc2[c]) * LN2 - g;
        }
    }
}

extern "C" void kernel_launch(void* const* d_in, const int* in_sizes, int n_in,
                              void* d_out, int out_size, void* d_ws, size_t ws_size,
                              hipStream_t stream) {
    const float* feats = (const float*)d_in[0];   // [B, L, 32] f32
    const float* trans = (const float*)d_in[1];   // [32, 32] f32
    const int*   tg    = (const int*)  d_in[2];   // [B, L] int
    float*       out   = (float*)d_out;           // [B] f32

    const int B = out_size;                       // 4096
    const int L = in_sizes[0] / (B * TAGS);       // 1024

    crf_nll_kernel<<<dim3(B / (NC * 16)), dim3(32), 0, stream>>>(feats, trans, tg, out, B, L);
}
